// AttentionAggregation_79774722556655
// MI455X (gfx1250) — matrix operand order repaired and checked by reading, not compiled, not run on hardware
//
#include <hip/hip_runtime.h>
#include <hip/hip_bf16.h>

typedef __attribute__((ext_vector_type(16))) __bf16 v16bf;
typedef __attribute__((ext_vector_type(8)))  __bf16 v8bf;
typedef __attribute__((ext_vector_type(8)))  float  v8f;
typedef __attribute__((ext_vector_type(4)))  int    v4i;

typedef __attribute__((address_space(1))) v4i* gv4i_p;  // global int4*
typedef __attribute__((address_space(3))) v4i* lv4i_p;  // LDS int4*

#define D_DIM   512
#define N_SEQ   1024
#define B_BATCH 64
#define BM      128
#define BN      128
#define BK      32
#define LDH     520   // H / Q tile LDS row stride (bf16 elems), keeps 16B alignment
#define LDT     40    // staging tile LDS row stride (bf16 elems)

#if __has_builtin(__builtin_amdgcn_global_load_async_to_lds_b128)
#define HAS_ASYNC_LDS 1
#else
#define HAS_ASYNC_LDS 0
#endif

// 16-byte global -> LDS copy. Prefer the CDNA5 async-to-LDS path (no VGPR
// round trip, tracked by ASYNCcnt); fall back to load+ds_store.
static __device__ __forceinline__ void cp16(const __bf16* g, __bf16* l) {
#if HAS_ASYNC_LDS
  __builtin_amdgcn_global_load_async_to_lds_b128((gv4i_p)g, (lv4i_p)l, 0, 0);
#else
  *(v8bf*)l = *(const v8bf*)g;
#endif
}

static __device__ __forceinline__ void async_wait() {
#if HAS_ASYNC_LDS
#if __has_builtin(__builtin_amdgcn_s_wait_asynccnt)
  __builtin_amdgcn_s_wait_asynccnt(0);
#else
  asm volatile("s_wait_asynccnt 0" ::: "memory");
#endif
#endif
}

// Build a 16x32 bf16 A-fragment per CDNA5 ISA layout:
// lane<16:  elems 0..7 = A[m][K0..K0+7],  elems 8..15 = A[m][K0+16..K0+23], K0 = 0
// lane>=16: same with K0 = 8  -> caller passes p = &A[m][half*8]
static __device__ __forceinline__ v16bf frag_a(const __bf16* p) {
  v8bf lo = *(const v8bf*)p;
  v8bf hi = *(const v8bf*)(p + 16);
  v16bf r;
#pragma unroll
  for (int i = 0; i < 8; ++i) { r[i] = lo[i]; r[i + 8] = hi[i]; }
  return r;
}

static __device__ __forceinline__ v8f wmma_bf16(v16bf a, v16bf b, v8f c) {
  // (neg_a, A, neg_b, B, c_mod, C, reuse_a, reuse_b)
  return __builtin_amdgcn_wmma_f32_16x16x32_bf16(false, a, false, b, (short)0, c,
                                                 false, false);
}

static __device__ __forceinline__ void zero_acc(v8f acc[8]) {
#pragma unroll
  for (int t = 0; t < 8; ++t)
#pragma unroll
    for (int i = 0; i < 8; ++i) acc[t][i] = 0.0f;
}

// ---------------------------------------------------------------------------
// Kernel 0: convert + transpose one 512x512 f32 weight -> bf16 W^T
// wt[e*D + d] = w[d*D + e]; B-fragments then read contiguous K runs.
// ---------------------------------------------------------------------------
__global__ __launch_bounds__(256) void cvt_transpose_kernel(
    const float* __restrict__ w, __bf16* __restrict__ wt) {
  int i = blockIdx.x * 256 + threadIdx.x;  // grid covers D*D
  int e = i & (D_DIM - 1);
  int d = i >> 9;
  wt[(size_t)e * D_DIM + d] = (__bf16)w[i];
}

// ---------------------------------------------------------------------------
// Kernel 1: fused 2-layer MLP, 128 rows per block.
//   H = relu(x @ W1 + b1)  (kept in LDS, bf16)
//   Y = H @ W2 + b2        (stored to global, bf16)
// ---------------------------------------------------------------------------
__global__ __launch_bounds__(256) void mlp_fused_kernel(
    const float* __restrict__ x, const __bf16* __restrict__ w1t,
    const float* __restrict__ b1, const __bf16* __restrict__ w2t,
    const float* __restrict__ b2, __bf16* __restrict__ y) {
  __shared__ __align__(16) __bf16 Hs[BM * LDH];
  __shared__ __align__(16) __bf16 Ast[BM * LDT];
  __shared__ __align__(16) __bf16 Bst[BN * LDT];

  const int tid  = threadIdx.x;
  const int lane = tid & 31;
  const int wave = tid >> 5;
  const int m15  = lane & 15;
  const int half = lane >> 4;
  const size_t row0 = (size_t)blockIdx.x * BM;
  const int sr = tid >> 1;           // staging row 0..127
  const int sc = (tid & 1) * 16;     // staging col half 0/16

  // -------- Phase 1: H = relu(x @ W1 + b1), in 128-column chunks --------
  for (int nc = 0; nc < 4; ++nc) {
    v8f acc[8];
    zero_acc(acc);
    for (int ks = 0; ks < 16; ++ks) {
      __syncthreads();
      {  // stage A tile: x[row0..+127][ks*32..+31], f32 -> bf16 (VALU convert)
        const float* gp = x + (row0 + sr) * D_DIM + ks * BK + sc;
        __bf16* dp = &Ast[sr * LDT + sc];
#pragma unroll
        for (int j = 0; j < 16; j += 4) {
          float4 v = *(const float4*)(gp + j);
          dp[j + 0] = (__bf16)v.x; dp[j + 1] = (__bf16)v.y;
          dp[j + 2] = (__bf16)v.z; dp[j + 3] = (__bf16)v.w;
        }
      }
      {  // stage B tile: W1^T rows (out-col n, K contiguous) -- async to LDS
        const __bf16* gp = w1t + (size_t)(nc * BN + sr) * D_DIM + ks * BK + sc;
        cp16(gp,     &Bst[sr * LDT + sc]);
        cp16(gp + 8, &Bst[sr * LDT + sc + 8]);
      }
      async_wait();
      __syncthreads();
      const __bf16* ap = &Ast[(wave * 16 + m15) * LDT + half * 8];
      v16bf a = frag_a(ap);
#pragma unroll
      for (int t = 0; t < 8; ++t) {
        v16bf b = frag_a(&Bst[(t * 16 + m15) * LDT + half * 8]);
        acc[t] = wmma_bf16(a, b, acc[t]);
      }
    }
    // epilogue: bias + relu -> Hs (bf16)
#pragma unroll
    for (int t = 0; t < 8; ++t) {
      const int col = nc * BN + t * 16 + m15;
      const float bias = b1[col];
#pragma unroll
      for (int r = 0; r < 8; ++r) {
        float v = acc[t][r] + bias;
        v = v > 0.0f ? v : 0.0f;
        Hs[(wave * 16 + half * 8 + r) * LDH + col] = (__bf16)v;
      }
    }
  }
  __syncthreads();

  // -------- Phase 2: Y = H @ W2 + b2 --------
  for (int nc = 0; nc < 4; ++nc) {
    v8f acc[8];
    zero_acc(acc);
    for (int ks = 0; ks < 16; ++ks) {
      __syncthreads();
      {  // stage B tile: W2^T -- async to LDS
        const __bf16* gp = w2t + (size_t)(nc * BN + sr) * D_DIM + ks * BK + sc;
        cp16(gp,     &Bst[sr * LDT + sc]);
        cp16(gp + 8, &Bst[sr * LDT + sc + 8]);
      }
      async_wait();
      __syncthreads();
      const __bf16* ap = &Hs[(wave * 16 + m15) * LDH + ks * BK + half * 8];
      v16bf a = frag_a(ap);
#pragma unroll
      for (int t = 0; t < 8; ++t) {
        v16bf b = frag_a(&Bst[(t * 16 + m15) * LDT + half * 8]);
        acc[t] = wmma_bf16(a, b, acc[t]);
      }
    }
#pragma unroll
    for (int t = 0; t < 8; ++t) {
      const int col = nc * BN + t * 16 + m15;
      const float bias = b2[col];
#pragma unroll
      for (int r = 0; r < 8; ++r) {
        y[(row0 + wave * 16 + half * 8 + r) * D_DIM + col] =
            (__bf16)(acc[t][r] + bias);
      }
    }
  }
}

// ---------------------------------------------------------------------------
// Kernel 2: attention column-sums.
// Per (batch, 128-query tile): two passes over key tiles.
//   pass 0: online row-max m_q and denominator Z_q of softmax(QK^T/sqrt(D))
//   pass 1: recompute S, p = exp(s-m)/Z, accumulate column sums into c[b,k]
// ---------------------------------------------------------------------------
__global__ __launch_bounds__(256) void attn_colsum_kernel(
    const __bf16* __restrict__ Q, const __bf16* __restrict__ Km,
    float* __restrict__ csum) {
  __shared__ __align__(16) __bf16 Qs[BM * LDH];
  __shared__ __align__(16) __bf16 Kst[BN * LDT];
  __shared__ float cl[N_SEQ];

  const int tid  = threadIdx.x;
  const int lane = tid & 31;
  const int wave = tid >> 5;
  const int m15  = lane & 15;
  const int half = lane >> 4;
  const int b    = blockIdx.y;
  const int q0   = blockIdx.x * BM;
  const int sr   = tid >> 1;
  const int sc   = (tid & 1) * 16;

  for (int i = tid; i < N_SEQ; i += 256) cl[i] = 0.0f;

  // Load 128x512 Q tile (bf16) into LDS via async copies
#pragma unroll
  for (int i = 0; i < 32; ++i) {
    int flat = i * 2048 + tid * 8;
    int r = flat >> 9;
    int col = flat & (D_DIM - 1);
    cp16(&Q[((size_t)(b * N_SEQ + q0 + r)) * D_DIM + col], &Qs[r * LDH + col]);
  }
  async_wait();

  const float scale = 0.04419417382415922f;  // 1/sqrt(512)
  float m_run[8], z_run[8];
#pragma unroll
  for (int r = 0; r < 8; ++r) { m_run[r] = -3.0e38f; z_run[r] = 0.0f; }

  for (int pass = 0; pass < 2; ++pass) {
    for (int kt = 0; kt < 8; ++kt) {
      v8f acc[8];
      zero_acc(acc);
      for (int ks = 0; ks < 16; ++ks) {
        __syncthreads();
        {  // stage key tile rows: K[key][d-chunk] == B^T rows (K contiguous)
          const __bf16* gp =
              Km + ((size_t)(b * N_SEQ + kt * BN + sr)) * D_DIM + ks * BK + sc;
          cp16(gp,     &Kst[sr * LDT + sc]);
          cp16(gp + 8, &Kst[sr * LDT + sc + 8]);
        }
        async_wait();
        __syncthreads();
        const __bf16* ap = &Qs[(wave * 16 + m15) * LDH + ks * BK + half * 8];
        v16bf a = frag_a(ap);
#pragma unroll
        for (int t = 0; t < 8; ++t) {
          v16bf bb = frag_a(&Kst[(t * 16 + m15) * LDT + half * 8]);
          acc[t] = wmma_bf16(a, bb, acc[t]);
        }
      }
      if (pass == 0) {
        // online max / sum-exp per query row (rows split across lane halves)
#pragma unroll
        for (int r = 0; r < 8; ++r) {
          float mx = -3.0e38f;
#pragma unroll
          for (int t = 0; t < 8; ++t) mx = fmaxf(mx, acc[t][r] * scale);
#pragma unroll
          for (int d = 1; d < 16; d <<= 1) mx = fmaxf(mx, __shfl_xor(mx, d, 32));
          const float mnew = fmaxf(m_run[r], mx);
          float s = 0.0f;
#pragma unroll
          for (int t = 0; t < 8; ++t) s += __expf(acc[t][r] * scale - mnew);
#pragma unroll
          for (int d = 1; d < 16; d <<= 1) s += __shfl_xor(s, d, 32);
          z_run[r] = z_run[r] * __expf(m_run[r] - mnew) + s;
          m_run[r] = mnew;
        }
      } else {
        float rz[8];
#pragma unroll
        for (int r = 0; r < 8; ++r) rz[r] = 1.0f / z_run[r];
#pragma unroll
        for (int t = 0; t < 8; ++t) {
          float cs = 0.0f;
#pragma unroll
          for (int r = 0; r < 8; ++r)
            cs += __expf(acc[t][r] * scale - m_run[r]) * rz[r];
          cs += __shfl_xor(cs, 16, 32);  // combine the two 8-row halves
          if (half == 0) atomicAdd(&cl[kt * BN + t * 16 + m15], cs);
        }
      }
    }
  }
  __syncthreads();
  for (int i = tid; i < N_SEQ; i += 256) atomicAdd(&csum[b * N_SEQ + i], cl[i]);
}

// ---------------------------------------------------------------------------
// Kernel 3: out[b,:] = (1/N) * sum_k c[b,k] * x[b,k,:]   (memory-bound GEMV)
// ---------------------------------------------------------------------------
__global__ __launch_bounds__(256) void out_gemv_kernel(
    const float* __restrict__ x, const float* __restrict__ csum,
    float* __restrict__ out) {
  __shared__ float cs[256];
  const int b = blockIdx.x;
  const int d = threadIdx.x;
  float a0 = 0.0f, a1 = 0.0f;
  for (int kc = 0; kc < 4; ++kc) {
    __syncthreads();
    cs[d] = csum[b * N_SEQ + kc * 256 + d];
    __syncthreads();
    for (int j = 0; j < 256; ++j) {
      const float* xp = x + ((size_t)(b * N_SEQ + kc * 256 + j)) * D_DIM;
      const float w = cs[j];
      a0 = fmaf(w, xp[d], a0);
      a1 = fmaf(w, xp[d + 256], a1);
    }
  }
  out[b * D_DIM + d]       = a0 * (1.0f / (float)N_SEQ);
  out[b * D_DIM + d + 256] = a1 * (1.0f / (float)N_SEQ);
}

// ---------------------------------------------------------------------------
extern "C" void kernel_launch(void* const* d_in, const int* in_sizes, int n_in,
                              void* d_out, int out_size, void* d_ws,
                              size_t ws_size, hipStream_t stream) {
  const float* x   = (const float*)d_in[0];
  const float* qw1 = (const float*)d_in[1];
  const float* qb1 = (const float*)d_in[2];
  const float* qw2 = (const float*)d_in[3];
  const float* qb2 = (const float*)d_in[4];
  const float* kw1 = (const float*)d_in[5];
  const float* kb1 = (const float*)d_in[6];
  const float* kw2 = (const float*)d_in[7];
  const float* kb2 = (const float*)d_in[8];
  float* out = (float*)d_out;

  char* ws = (char*)d_ws;
  const size_t QKB = (size_t)B_BATCH * N_SEQ * D_DIM * sizeof(__bf16);  // 64 MiB
  const size_t WB  = (size_t)D_DIM * D_DIM * sizeof(__bf16);            // 512 KiB
  __bf16* qbuf = (__bf16*)(ws);
  __bf16* kbuf = (__bf16*)(ws + QKB);
  __bf16* w1tq = (__bf16*)(ws + 2 * QKB);
  __bf16* w2tq = (__bf16*)(ws + 2 * QKB + WB);
  __bf16* w1tk = (__bf16*)(ws + 2 * QKB + 2 * WB);
  __bf16* w2tk = (__bf16*)(ws + 2 * QKB + 3 * WB);
  float*  csum = (float*)(ws + 2 * QKB + 4 * WB);

  const int cvtGrid = (D_DIM * D_DIM) / 256;
  cvt_transpose_kernel<<<cvtGrid, 256, 0, stream>>>(qw1, w1tq);
  cvt_transpose_kernel<<<cvtGrid, 256, 0, stream>>>(qw2, w2tq);
  cvt_transpose_kernel<<<cvtGrid, 256, 0, stream>>>(kw1, w1tk);
  cvt_transpose_kernel<<<cvtGrid, 256, 0, stream>>>(kw2, w2tk);

  const int rows = B_BATCH * N_SEQ;
  mlp_fused_kernel<<<rows / BM, 256, 0, stream>>>(x, w1tq, qb1, w2tq, qb2, qbuf);
  mlp_fused_kernel<<<rows / BM, 256, 0, stream>>>(x, w1tk, kb1, w2tk, kb2, kbuf);

  (void)hipMemsetAsync(csum, 0, (size_t)B_BATCH * N_SEQ * sizeof(float), stream);
  attn_colsum_kernel<<<dim3(N_SEQ / BM, B_BATCH), 256, 0, stream>>>(qbuf, kbuf,
                                                                    csum);
  out_gemv_kernel<<<B_BATCH, 256, 0, stream>>>(x, csum, out);

  (void)in_sizes; (void)n_in; (void)out_size; (void)ws_size;
}
